// ZImageAttention_36369783062541
// MI455X (gfx1250) — compile-verified, matrix operand order checked by reading, not run on hardware
//
#include <hip/hip_runtime.h>

// ---------------------------------------------------------------------------
// Problem constants (from the reference)
// ---------------------------------------------------------------------------
#define DIM_    3072
#define NHEADS  24
#define HD      128
#define BB      2
#define SS      2048
#define MROWS   (BB * SS)      // 4096
#define NQKV    (3 * DIM_)     // 9216
#define EPS     1e-6f
#define SCALE_ATTN 0.08838834764831845f   // 1/sqrt(128)

// ---------------------------------------------------------------------------
// Types for WMMA (gfx1250, wave32): v_wmma_f32_16x16x32_bf16
//   A: 16x32 bf16  (lane = M%16, K chunks at kbase=(lane/16)*8 and kbase+16)
//   B: 32x16 bf16  (lane = N%16, same K chunking)
//   C/D: 16x16 f32 (lane = N%16, VGPR v -> M = v + 8*(lane/16))
// ---------------------------------------------------------------------------
typedef __attribute__((ext_vector_type(16))) __bf16         v16bf;
typedef __attribute__((ext_vector_type(8)))  float          v8f;
typedef __attribute__((ext_vector_type(8)))  unsigned short u16x8;
typedef __attribute__((ext_vector_type(4)))  unsigned int   v4u;
typedef __attribute__((ext_vector_type(4)))  int            v4i;
typedef __attribute__((ext_vector_type(8)))  int            v8i;

union Frag {
    v16bf bf;
    u16x8 u[2];
};

__device__ __forceinline__ unsigned short f2bf(float f) {
    unsigned int u = __float_as_uint(f);
    u += 0x7FFFu + ((u >> 16) & 1u);         // round-to-nearest-even
    return (unsigned short)(u >> 16);
}
__device__ __forceinline__ float bf2f(unsigned short u) {
    return __uint_as_float(((unsigned int)u) << 16);
}

__device__ __forceinline__ v8f wmma_bf16(const Frag& a, const Frag& b, v8f c) {
    return __builtin_amdgcn_wmma_f32_16x16x32_bf16(
        false, a.bf, false, b.bf, (short)0, c, false, false);
}

// Load a 16x32 bf16 operand fragment from row-major memory (stride in elems).
__device__ __forceinline__ Frag load_frag_g(const unsigned short* __restrict__ p,
                                            int stride) {
    int lane = threadIdx.x & 31;
    int r    = lane & 15;
    int kb   = (lane >> 4) << 3;
    const unsigned short* q = p + (size_t)r * stride + kb;
    Frag f;
    f.u[0] = *(const u16x8*)(q);
    f.u[1] = *(const u16x8*)(q + 16);
    return f;
}

// ---------------------------------------------------------------------------
// TDM (Tensor Data Mover) availability
// ---------------------------------------------------------------------------
#if defined(__has_builtin)
#  if __has_builtin(__builtin_amdgcn_tensor_load_to_lds)
#    define HAVE_TDM 1
#  endif
#endif
#ifndef HAVE_TDM
#  define HAVE_TDM 0
#endif

#define LDSS 40   // padded LDS row stride (halfwords): conflict-free b128 reads

#if HAVE_TDM
// Issue one TENSOR_LOAD_TO_LDS moving a 128(row) x 32(bf16) tile from a
// row-major bf16 matrix (row stride K elems) into LDS, inserting 4 DWORDs of
// pad every 16 DWORDs so the LDS row stride becomes LDSS=40 halfwords.
// Must be executed by exactly one wave of the block (EXEC is ignored by TDM).
__device__ __forceinline__ void tdm_load_tile(const unsigned short* gsrc,
                                              unsigned lds_byte_addr, int K) {
    unsigned long long ga = (unsigned long long)(size_t)gsrc;
    v4u g0;
    g0[0] = 1u;                                    // count=1 (valid descriptor)
    g0[1] = lds_byte_addr;                         // lds_addr
    g0[2] = (unsigned)(ga & 0xFFFFFFFFull);        // global_addr[31:0]
    g0[3] = (unsigned)((ga >> 32) & 0x01FFFFFFull) // global_addr[56:32]
            | (2u << 30);                          // type=2 ("image")
    v8i g1;
    // data_size=1(2B) | pad_enable | pad_interval=3(16 DW) | pad_amount=3(4 DW)
    g1[0] = (int)((1u << 16) | (1u << 20) | (3u << 22) | (3u << 25));
    g1[1] = (int)(((unsigned)K & 0xFFFFu) << 16);  // tensor_dim0 lo16 @[63:48]
    g1[2] = (int)(((unsigned)K >> 16)              // tensor_dim0 hi16 @[79:64]
            | (128u << 16));                       // tensor_dim1 lo16 @[95:80]
    g1[3] = (int)(32u << 16);                      // tensor_dim1 hi=0, tile_dim0=32
    g1[4] = (int)128u;                             // tile_dim1=128, tile_dim2=0
    g1[5] = (int)K;                                // tensor_dim0_stride lo32
    g1[6] = 0;                                     // stride0 hi16, stride1 lo16
    g1[7] = 0;
    v4i g2 = {0, 0, 0, 0};
    v4i g3 = {0, 0, 0, 0};
#if __clang_major__ >= 23
    v8i g4 = {0, 0, 0, 0, 0, 0, 0, 0};
    __builtin_amdgcn_tensor_load_to_lds(g0, g1, g2, g3, g4, 0);
#else
    __builtin_amdgcn_tensor_load_to_lds(g0, g1, g2, g3, 0);
#endif
}

__device__ __forceinline__ void tdm_wait0() {
#if defined(__has_builtin) && __has_builtin(__builtin_amdgcn_s_wait_tensorcnt)
    __builtin_amdgcn_s_wait_tensorcnt(0);
#else
    asm volatile("s_wait_tensorcnt 0" ::: "memory");
#endif
}
#endif // HAVE_TDM

// ---------------------------------------------------------------------------
// f32 -> bf16 bulk conversion (weights): 8 elements/thread
// ---------------------------------------------------------------------------
__global__ void __launch_bounds__(256)
cvt_f32_bf16_kernel(const float* __restrict__ src,
                    unsigned short* __restrict__ dst, long long n) {
    long long i = ((long long)blockIdx.x * 256 + threadIdx.x) * 8;
    if (i + 8 <= n) {
        const float4* s4 = (const float4*)(src + i);
        float4 a = s4[0], b = s4[1];
        u16x8 o;
        o[0] = f2bf(a.x); o[1] = f2bf(a.y); o[2] = f2bf(a.z); o[3] = f2bf(a.w);
        o[4] = f2bf(b.x); o[5] = f2bf(b.y); o[6] = f2bf(b.z); o[7] = f2bf(b.w);
        *(u16x8*)(dst + i) = o;
    }
}

// ---------------------------------------------------------------------------
// Kernel: tiled bf16 WMMA GEMM,  C[M,N] = A[M,K] * B[N,K]^T
//   AT = float  -> convert f32->bf16 while staging A into LDS
//   AT = ushort -> A already bf16
//   B is bf16; its tile is staged by the Tensor Data Mover (wave 0 issues the
//   DMA, waits TENSORcnt==0, then the workgroup barrier publishes the tile).
// Block: 256 threads (8 waves). Block tile 128x128, wave tile 32x64.
// ---------------------------------------------------------------------------
template <typename AT>
__device__ __forceinline__ void stage_tile(const AT* __restrict__ g, int ld,
                                           int row0, int k0,
                                           unsigned short* __restrict__ sh) {
    int t    = threadIdx.x;
    int row  = t >> 1;
    int half = (t & 1) << 4;                 // 0 or 16 elements
    const AT* src = g + (size_t)(row0 + row) * ld + k0 + half;
    unsigned short* dst = sh + row * LDSS + half;
    if constexpr (sizeof(AT) == 4) {         // f32 -> bf16 convert
        const float4* s4 = (const float4*)src;
        float4 f0 = s4[0], f1 = s4[1], f2 = s4[2], f3 = s4[3];
        u16x8 o0, o1;
        o0[0] = f2bf(f0.x); o0[1] = f2bf(f0.y); o0[2] = f2bf(f0.z); o0[3] = f2bf(f0.w);
        o0[4] = f2bf(f1.x); o0[5] = f2bf(f1.y); o0[6] = f2bf(f1.z); o0[7] = f2bf(f1.w);
        o1[0] = f2bf(f2.x); o1[1] = f2bf(f2.y); o1[2] = f2bf(f2.z); o1[3] = f2bf(f2.w);
        o1[4] = f2bf(f3.x); o1[5] = f2bf(f3.y); o1[6] = f2bf(f3.z); o1[7] = f2bf(f3.w);
        *(u16x8*)(dst)     = o0;
        *(u16x8*)(dst + 8) = o1;
    } else {                                  // already bf16
        const u16x8* s8 = (const u16x8*)src;
        *(u16x8*)(dst)     = s8[0];
        *(u16x8*)(dst + 8) = s8[1];
    }
}

template <typename AT, bool OUT_BF16>
__global__ void __launch_bounds__(256)
gemm_wmma_kernel(const AT* __restrict__ A, const unsigned short* __restrict__ Bw,
                 void* __restrict__ Cout, int M, int N, int K) {
    __shared__ unsigned short shA[128 * LDSS];
    __shared__ unsigned short shB[128 * LDSS];

    const int m0   = blockIdx.y * 128;
    const int n0   = blockIdx.x * 128;
    const int wid  = threadIdx.x >> 5;
    const int lane = threadIdx.x & 31;
    const int wm   = (wid & 3) * 32;      // 4 waves along M
    const int wn   = (wid >> 2) * 64;     // 2 waves along N
    const int r    = lane & 15;
    const int kb   = (lane >> 4) << 3;
    const int hl   = lane >> 4;

#if HAVE_TDM
    const unsigned shB_addr = (unsigned)(size_t)(void*)&shB[0];
#endif

    v8f acc[2][4];
#pragma unroll
    for (int im = 0; im < 2; ++im)
#pragma unroll
        for (int in = 0; in < 4; ++in)
#pragma unroll
            for (int v = 0; v < 8; ++v) acc[im][in][v] = 0.0f;

    for (int k0 = 0; k0 < K; k0 += 32) {
#if HAVE_TDM
        if (wid == 0)                         // one wave issues the tensor DMA
            tdm_load_tile(Bw + (size_t)n0 * K + k0, shB_addr, K);
        stage_tile<AT>(A, K, m0, k0, shA);
        if (wid == 0) tdm_wait0();            // DMA done before the barrier
#else
        stage_tile<AT>(A, K, m0, k0, shA);
        stage_tile<unsigned short>(Bw, K, n0, k0, shB);
#endif
        __syncthreads();

        Frag aF[2], bF[4];
#pragma unroll
        for (int im = 0; im < 2; ++im) {
            int rr = wm + im * 16 + r;
            aF[im].u[0] = *(const u16x8*)&shA[rr * LDSS + kb];
            aF[im].u[1] = *(const u16x8*)&shA[rr * LDSS + kb + 16];
        }
#pragma unroll
        for (int in = 0; in < 4; ++in) {
            int rr = wn + in * 16 + r;
            bF[in].u[0] = *(const u16x8*)&shB[rr * LDSS + kb];
            bF[in].u[1] = *(const u16x8*)&shB[rr * LDSS + kb + 16];
        }
#pragma unroll
        for (int im = 0; im < 2; ++im)
#pragma unroll
            for (int in = 0; in < 4; ++in)
                acc[im][in] = wmma_bf16(aF[im], bF[in], acc[im][in]);
        __syncthreads();
    }

#pragma unroll
    for (int im = 0; im < 2; ++im)
#pragma unroll
        for (int in = 0; in < 4; ++in) {
            int col = n0 + wn + in * 16 + r;
#pragma unroll
            for (int v = 0; v < 8; ++v) {
                int row = m0 + wm + im * 16 + v + 8 * hl;
                if constexpr (OUT_BF16)
                    ((unsigned short*)Cout)[(size_t)row * N + col] = f2bf(acc[im][in][v]);
                else
                    ((float*)Cout)[(size_t)row * N + col] = acc[im][in][v];
            }
        }
}

// ---------------------------------------------------------------------------
// Per-head RMSNorm + interleaved RoPE on Q,K; re-layout Q,K to (b,h,s,d) and
// V transposed to (b,h,d,s). One block per (b,s).
// ---------------------------------------------------------------------------
__device__ __forceinline__ void qk_head(const unsigned short* __restrict__ src,
                                        const float* __restrict__ wn,
                                        float c0, float s0, float c1, float s1,
                                        int d0, unsigned short* __restrict__ dst) {
    float x0 = bf2f(src[0]), x1 = bf2f(src[1]);
    float x2 = bf2f(src[2]), x3 = bf2f(src[3]);
    float sq = x0 * x0 + x1 * x1 + x2 * x2 + x3 * x3;
#pragma unroll
    for (int off = 16; off > 0; off >>= 1) sq += __shfl_xor(sq, off, 32);
    float rms = rsqrtf(sq * (1.0f / (float)HD) + EPS);
    x0 *= rms * wn[d0 + 0]; x1 *= rms * wn[d0 + 1];
    x2 *= rms * wn[d0 + 2]; x3 *= rms * wn[d0 + 3];
    float o0 = x0 * c0 - x1 * s0, o1 = x1 * c0 + x0 * s0;
    float o2 = x2 * c1 - x3 * s1, o3 = x3 * c1 + x2 * s1;
    dst[0] = f2bf(o0); dst[1] = f2bf(o1); dst[2] = f2bf(o2); dst[3] = f2bf(o3);
}

__global__ void __launch_bounds__(256)
norm_rope_kernel(const unsigned short* __restrict__ qkv,
                 const float* __restrict__ cosb, const float* __restrict__ sinb,
                 const float* __restrict__ wq, const float* __restrict__ wk,
                 unsigned short* __restrict__ Qb, unsigned short* __restrict__ Kb,
                 unsigned short* __restrict__ Vt) {
    const int s    = blockIdx.x;
    const int b    = blockIdx.y;
    const int wid  = threadIdx.x >> 5;
    const int lane = threadIdx.x & 31;
    const int d0   = lane * 4;
    const int i0   = lane * 2;

    const float c0 = cosb[s * (HD / 2) + i0],     sn0 = sinb[s * (HD / 2) + i0];
    const float c1 = cosb[s * (HD / 2) + i0 + 1], sn1 = sinb[s * (HD / 2) + i0 + 1];

    const size_t base = (size_t)(b * SS + s) * (size_t)NQKV;

    for (int h = wid; h < NHEADS; h += 8) {
        const size_t bh = (size_t)(b * NHEADS + h);
        qk_head(qkv + base + h * HD + d0, wq, c0, sn0, c1, sn1, d0,
                Qb + (bh * SS + s) * HD + d0);
        qk_head(qkv + base + DIM_ + h * HD + d0, wk, c0, sn0, c1, sn1, d0,
                Kb + (bh * SS + s) * HD + d0);
        const unsigned short* vsrc = qkv + base + 2 * DIM_ + h * HD + d0;
#pragma unroll
        for (int j = 0; j < 4; ++j)
            Vt[(bh * HD + d0 + j) * SS + s] = vsrc[j];
    }
}

// ---------------------------------------------------------------------------
// Flash attention. Block = 4 waves (128 threads); each wave owns 16 Q rows x
// full head dim; online softmax over K/V in 32-row steps.
// ---------------------------------------------------------------------------
__global__ void __launch_bounds__(128)
flash_attn_kernel(const unsigned short* __restrict__ Qb,
                  const unsigned short* __restrict__ Kb,
                  const unsigned short* __restrict__ Vt,
                  unsigned short* __restrict__ attn) {
    __shared__ unsigned short shP[4 * 16 * LDSS];   // per-wave P staging

    const int q0   = blockIdx.x * 64;
    const int h    = blockIdx.y;
    const int b    = blockIdx.z;
    const int wid  = threadIdx.x >> 5;
    const int lane = threadIdx.x & 31;
    const int r    = lane & 15;
    const int kb   = (lane >> 4) << 3;
    const int hl   = lane >> 4;
    const size_t bh = (size_t)(b * NHEADS + h);

    Frag qf[4];
    const unsigned short* qptr = Qb + (bh * SS + q0 + wid * 16) * HD;
#pragma unroll
    for (int kk = 0; kk < 4; ++kk) qf[kk] = load_frag_g(qptr + kk * 32, HD);

    v8f o[8];
    float m_i[8], l_i[8];
#pragma unroll
    for (int n = 0; n < 8; ++n)
#pragma unroll
        for (int v = 0; v < 8; ++v) o[n][v] = 0.0f;
#pragma unroll
    for (int v = 0; v < 8; ++v) { m_i[v] = -1e30f; l_i[v] = 0.0f; }

    unsigned short* pS = &shP[wid * 16 * LDSS];

    for (int k0 = 0; k0 < SS; k0 += 32) {
        v8f sf[2];
#pragma unroll
        for (int f = 0; f < 2; ++f)
#pragma unroll
            for (int v = 0; v < 8; ++v) sf[f][v] = 0.0f;

        const unsigned short* kptr = Kb + (bh * SS + k0) * HD;
#pragma unroll
        for (int kk = 0; kk < 4; ++kk) {
            Frag kf0 = load_frag_g(kptr + kk * 32, HD);
            Frag kf1 = load_frag_g(kptr + 16 * HD + kk * 32, HD);
            sf[0] = wmma_bf16(qf[kk], kf0, sf[0]);
            sf[1] = wmma_bf16(qf[kk], kf1, sf[1]);
        }

        float mnew[8], alpha[8];
#pragma unroll
        for (int v = 0; v < 8; ++v) {
            sf[0][v] *= SCALE_ATTN;
            sf[1][v] *= SCALE_ATTN;
            float mv = fmaxf(sf[0][v], sf[1][v]);
#pragma unroll
            for (int off = 1; off < 16; off <<= 1)
                mv = fmaxf(mv, __shfl_xor(mv, off, 32));
            mnew[v]  = fmaxf(m_i[v], mv);
            alpha[v] = __expf(m_i[v] - mnew[v]);
        }
#pragma unroll
        for (int v = 0; v < 8; ++v) {
            sf[0][v] = __expf(sf[0][v] - mnew[v]);
            sf[1][v] = __expf(sf[1][v] - mnew[v]);
            float sv = sf[0][v] + sf[1][v];
#pragma unroll
            for (int off = 1; off < 16; off <<= 1)
                sv += __shfl_xor(sv, off, 32);
            l_i[v] = l_i[v] * alpha[v] + sv;
            m_i[v] = mnew[v];
        }
#pragma unroll
        for (int n = 0; n < 8; ++n)
#pragma unroll
            for (int v = 0; v < 8; ++v) o[n][v] *= alpha[v];

        // P (C-layout f32) -> bf16 A-fragment via per-wave LDS round-trip
#pragma unroll
        for (int f = 0; f < 2; ++f)
#pragma unroll
            for (int v = 0; v < 8; ++v)
                pS[(v + 8 * hl) * LDSS + f * 16 + r] = f2bf(sf[f][v]);
        asm volatile("s_wait_dscnt 0" ::: "memory");
        Frag pf;
        pf.u[0] = *(const u16x8*)&pS[r * LDSS + kb];
        pf.u[1] = *(const u16x8*)&pS[r * LDSS + kb + 16];

#pragma unroll
        for (int n = 0; n < 8; ++n) {
            Frag vf = load_frag_g(Vt + (bh * HD + n * 16) * SS + k0, SS);
            o[n] = wmma_bf16(pf, vf, o[n]);
        }
    }

    float inv[8];
#pragma unroll
    for (int v = 0; v < 8; ++v) inv[v] = 1.0f / l_i[v];
    const size_t obase = (size_t)(b * SS + q0 + wid * 16) * DIM_ + (size_t)h * HD;
#pragma unroll
    for (int n = 0; n < 8; ++n) {
        int col = n * 16 + r;
#pragma unroll
        for (int v = 0; v < 8; ++v) {
            int row = v + 8 * hl;
            attn[obase + (size_t)row * DIM_ + col] = f2bf(o[n][v] * inv[v]);
        }
    }
}

// ---------------------------------------------------------------------------
// Launcher
// ---------------------------------------------------------------------------
extern "C" void kernel_launch(void* const* d_in, const int* in_sizes, int n_in,
                              void* d_out, int out_size, void* d_ws, size_t ws_size,
                              hipStream_t stream) {
    const float* hidden = (const float*)d_in[0];
    const float* cosb   = (const float*)d_in[1];
    const float* sinb   = (const float*)d_in[2];
    const float* w_qkv  = (const float*)d_in[3];
    const float* w_qn   = (const float*)d_in[4];
    const float* w_kn   = (const float*)d_in[5];
    const float* w_out  = (const float*)d_in[6];
    float* out = (float*)d_out;

    // Workspace layout (bf16 halfwords)
    unsigned short* qkv   = (unsigned short*)d_ws;                        // 4096 x 9216
    unsigned short* Qb    = qkv  + (size_t)MROWS * NQKV;                  // (b,h,s,d)
    unsigned short* Kb    = Qb   + (size_t)BB * NHEADS * SS * HD;         // (b,h,s,d)
    unsigned short* Vt    = Kb   + (size_t)BB * NHEADS * SS * HD;         // (b,h,d,s)
    unsigned short* attn  = Vt   + (size_t)BB * NHEADS * SS * HD;         // 4096 x 3072
    unsigned short* wqkvb = attn + (size_t)MROWS * DIM_;                  // 9216 x 3072
    unsigned short* woutb = wqkvb + (size_t)NQKV * DIM_;                  // 3072 x 3072

    // 0) weights f32 -> bf16 (enables TDM staging of B tiles)
    {
        long long n1 = (long long)NQKV * DIM_;
        long long n2 = (long long)DIM_ * DIM_;
        cvt_f32_bf16_kernel<<<(unsigned)(n1 / 8 / 256), 256, 0, stream>>>(w_qkv, wqkvb, n1);
        cvt_f32_bf16_kernel<<<(unsigned)(n2 / 8 / 256), 256, 0, stream>>>(w_out, woutb, n2);
    }

    // 1) QKV projection: qkv = hidden * w_qkv^T  (f32 A, bf16 B, bf16 out)
    gemm_wmma_kernel<float, true>
        <<<dim3(NQKV / 128, MROWS / 128), 256, 0, stream>>>(
            hidden, wqkvb, qkv, MROWS, NQKV, DIM_);

    // 2) per-head RMSNorm + RoPE on Q,K; V transpose
    norm_rope_kernel<<<dim3(SS, BB), 256, 0, stream>>>(
        qkv, cosb, sinb, w_qn, w_kn, Qb, Kb, Vt);

    // 3) flash attention
    flash_attn_kernel<<<dim3(SS / 64, NHEADS, BB), 128, 0, stream>>>(
        Qb, Kb, Vt, attn);

    // 4) output projection: out = attn * w_out^T  (bf16 A, bf16 B, f32 out)
    gemm_wmma_kernel<unsigned short, false>
        <<<dim3(DIM_ / 128, MROWS / 128), 256, 0, stream>>>(
            attn, woutb, out, MROWS, DIM_, DIM_);
}
